// SimpleVectorQuantizer_41154376630734
// MI455X (gfx1250) — matrix operand order repaired and checked by reading, not compiled, not run on hardware
//
#include <hip/hip_runtime.h>
#include <hip/hip_bf16.h>
#include <stdint.h>

// ---------------------------------------------------------------------------
// VQ nearest-code search, fused GEMM + argmin + gather, bf16x3 WMMA (gfx1250)
// Round 2: 6 independent WMMA accumulator chains (beats the 5-slot bf16
// WMMA->WMMA RAW hazard) + async global->LDS B-tile prefetch (ASYNCcnt).
// ---------------------------------------------------------------------------

typedef __bf16 v16bf __attribute__((ext_vector_type(16)));
typedef float  v8f   __attribute__((ext_vector_type(8)));

#define NB     64
#define DIM    256
#define KEMB   1024
#define HW     1024              // 32*32
#define NROWS  (NB * HW)         // 65536
#define TILE_M 128               // rows per block (8 waves x 16 rows)
#define NKT    (KEMB / 16)       // 64 column tiles
#define NDB    (DIM / 32)        // 8 K-blocks of 32 along the reduction dim

union BFrag {
  v16bf    v;
  uint32_t u[8];
  uint4    q[2];
};

// Split f into bf16 hi (RNE) + bf16 lo (RNE of exact residual).
__device__ __forceinline__ void split_bf16(float f, uint16_t& h, uint16_t& l) {
  uint32_t u  = __builtin_bit_cast(uint32_t, f);
  uint32_t r  = (u + 0x7FFFu + ((u >> 16) & 1u)) & 0xFFFF0000u;
  h           = (uint16_t)(r >> 16);
  float    lo = f - __builtin_bit_cast(float, r);
  uint32_t ul = __builtin_bit_cast(uint32_t, lo);
  l           = (uint16_t)((ul + 0x7FFFu + ((ul >> 16) & 1u)) >> 16);
}

// Async DMA of one B k-tile half (hi+lo fragments) into LDS.
// GVS addressing: lds-addr VGPR, 32-bit byte-offset VGPR, 64-bit SGPR base.
// The offset: immediate applies to BOTH the LDS and global side (ISA 08 §4.4),
// which matches our layout (LDS and global advance together within a region).
__device__ __forceinline__ void async_copy_b_tile(uint32_t lds_hi, uint32_t lds_lo,
                                                  uint32_t voff,
                                                  const uint32_t* gh,
                                                  const uint32_t* gl) {
  asm volatile(
      "global_load_async_to_lds_b128 %0, %2, %3 offset:0\n\t"
      "global_load_async_to_lds_b128 %0, %2, %3 offset:16\n\t"
      "global_load_async_to_lds_b128 %1, %2, %4 offset:0\n\t"
      "global_load_async_to_lds_b128 %1, %2, %4 offset:16"
      :
      : "v"(lds_hi), "v"(lds_lo), "v"(voff), "s"(gh), "s"(gl)
      : "memory");
}

__device__ __forceinline__ void wait_async_all() {
  asm volatile("s_wait_asynccnt 0x0" ::: "memory");
}

// -------- prep 1: eT[k][d] = e[d][k]; halfnorm[k] = 0.5*sum_d e[d][k]^2 -----
__global__ void vq_prep_norm_t(const float* __restrict__ e,
                               float* __restrict__ eT,
                               float* __restrict__ halfnorm) {
  int k = blockIdx.x * blockDim.x + threadIdx.x;
  if (k >= KEMB) return;
  float s = 0.f;
  for (int d = 0; d < DIM; ++d) {
    float v = e[(size_t)d * KEMB + k];
    eT[(size_t)k * DIM + d] = v;
    s += v * v;
  }
  halfnorm[k] = 0.5f * s;
}

// -------- prep 2: pre-swizzle codebook into WMMA B fragments (bf16 hi/lo) ---
// Layout: dword index = ((kt*NDB + db)*32 + lane)*8 + j
// lane<16: col = kt*16+lane, K rows (db*32 + 0..15);  lane>=16: K rows +16.
// VGPR j packs K = kb+2j (lo 16b) and kb+2j+1 (hi 16b).
__global__ void vq_prep_frags(const float* __restrict__ e,
                              uint32_t* __restrict__ bhi,
                              uint32_t* __restrict__ blo) {
  int t = blockIdx.x * blockDim.x + threadIdx.x;
  if (t >= NKT * NDB * 32) return;
  int lane = t & 31, db = (t >> 5) & 7, kt = t >> 8;
  int col = kt * 16 + (lane & 15);
  int kb  = db * 32 + ((lane & 16) ? 16 : 0);
  uint32_t hi[8], lo[8];
#pragma unroll
  for (int j = 0; j < 8; ++j) {
    uint16_t h0, l0, h1, l1;
    split_bf16(e[(size_t)(kb + 2 * j)     * KEMB + col], h0, l0);
    split_bf16(e[(size_t)(kb + 2 * j + 1) * KEMB + col], h1, l1);
    hi[j] = (uint32_t)h0 | ((uint32_t)h1 << 16);
    lo[j] = (uint32_t)l0 | ((uint32_t)l1 << 16);
  }
#pragma unroll
  for (int j = 0; j < 8; ++j) {
    bhi[(size_t)t * 8 + j] = hi[j];
    blo[(size_t)t * 8 + j] = lo[j];
  }
}

// -------- main: fused GEMM (bf16x3 WMMA) + argmin + dual quantized write ----
__global__ __launch_bounds__(256) void vq_main(
    const float* __restrict__ x,
    const uint32_t* __restrict__ bhi,
    const uint32_t* __restrict__ blo,
    const float* __restrict__ eT,
    const float* __restrict__ halfnorm,
    float* __restrict__ qout1,
    float* __restrict__ qout2,
    int* __restrict__ idx_out) {
  __shared__ __align__(16) uint32_t lds_b[2][2 * NDB * 32 * 8];  // 32 KB, dbl-buf
  __shared__ int lds_idx[8][16];

  const int tid  = threadIdx.x;
  const int lane = tid & 31;
  const int w    = tid >> 5;          // wave id 0..7
  const int mrow = lane & 15;         // row within 16-row wave tile
  const int half = lane >> 4;         // 0: lanes 0-15, 1: lanes 16-31
  const int n0   = blockIdx.x * TILE_M;
  const int b    = n0 >> 10;          // batch index (TILE_M divides HW)
  const int hw_row = (n0 & 1023) + w * 16 + mrow;
  const size_t xbase = (size_t)b * DIM * HW;

  // Per-thread DMA source/dest offsets for B staging (16B granules).
  const uint32_t stage_voff = (uint32_t)tid * 32u;

  // Build register-resident A fragments (bf16 hi/lo) in contiguous k order (k = 16*half + s), matching B.
  BFrag ahi[NDB], alo[NDB];
#pragma unroll
  for (int db = 0; db < NDB; ++db) {
#pragma unroll
    for (int j = 0; j < 8; ++j) {
      const int d0 = db * 32 + half * 16 + 2 * j;
      const float v0 = x[xbase + (size_t)d0 * HW + hw_row];
      const float v1 = x[xbase + (size_t)(d0 + 1) * HW + hw_row];
      uint16_t h0, l0, h1, l1;
      split_bf16(v0, h0, l0);
      split_bf16(v1, h1, l1);
      ahi[db].u[j] = (uint32_t)h0 | ((uint32_t)h1 << 16);
      alo[db].u[j] = (uint32_t)l0 | ((uint32_t)l1 << 16);
    }
  }

  float best[8];
  int   bidx[8];
#pragma unroll
  for (int r = 0; r < 8; ++r) { best[r] = -3.4e38f; bidx[r] = 0; }

  // Prime stage 0 of the B double buffer via async DMA.
  {
    const uint32_t lds_hi = (uint32_t)(uintptr_t)&lds_b[0][tid * 8];
    async_copy_b_tile(lds_hi, lds_hi + 8192u, stage_voff, bhi, blo);
    wait_async_all();
  }
  __syncthreads();

  for (int kt = 0; kt < NKT; ++kt) {
    const int cur = kt & 1;
    if (kt + 1 < NKT) {  // async-prefetch next B tile into the other buffer
      const uint32_t lds_hi = (uint32_t)(uintptr_t)&lds_b[cur ^ 1][tid * 8];
      async_copy_b_tile(lds_hi, lds_hi + 8192u, stage_voff,
                        bhi + (size_t)(kt + 1) * 2048,
                        blo + (size_t)(kt + 1) * 2048);
    }

    // 6 independent accumulator chains: 3 split terms x db-parity.
    // Same-chain WMMAs are 6 instructions apart -> outside the 5-slot
    // bf16 WMMA->WMMA RAW hazard window.
    v8f acc[6];
#pragma unroll
    for (int i = 0; i < 6; ++i) acc[i] = (v8f){0.f,0.f,0.f,0.f,0.f,0.f,0.f,0.f};
#pragma unroll
    for (int db = 0; db < NDB; ++db) {
      BFrag bh, bl;
      const uint4* ph = (const uint4*)&lds_b[cur][db * 256 + lane * 8];
      const uint4* pl = (const uint4*)&lds_b[cur][2048 + db * 256 + lane * 8];
      bh.q[0] = ph[0]; bh.q[1] = ph[1];
      bl.q[0] = pl[0]; bl.q[1] = pl[1];
      const int p = (db & 1) * 3;
      // sim ~= hi*hi + lo*hi + hi*lo  (fp32 accumulate)
      acc[p + 0] = __builtin_amdgcn_wmma_f32_16x16x32_bf16(
          false, ahi[db].v, false, bh.v, (short)0, acc[p + 0], false, false);
      acc[p + 1] = __builtin_amdgcn_wmma_f32_16x16x32_bf16(
          false, alo[db].v, false, bh.v, (short)0, acc[p + 1], false, false);
      acc[p + 2] = __builtin_amdgcn_wmma_f32_16x16x32_bf16(
          false, ahi[db].v, false, bl.v, (short)0, acc[p + 2], false, false);
    }

    // argmin(dist) == argmax(sim - 0.5*||e||^2); lane's column fixed per kt
    const int   col = kt * 16 + mrow;
    const float hn  = halfnorm[col];
#pragma unroll
    for (int r = 0; r < 8; ++r) {
      const float s = ((acc[0][r] + acc[3][r]) + (acc[1][r] + acc[4][r])) +
                      (acc[2][r] + acc[5][r]) - hn;
      if (s > best[r]) { best[r] = s; bidx[r] = col; }
    }

    wait_async_all();   // drain the prefetch (overlapped with the 24 WMMAs)
    __syncthreads();
  }

  // Cross-lane reduce: each row's candidates live in 16 lanes of one half.
#pragma unroll
  for (int r = 0; r < 8; ++r) {
    float bv = best[r];
    int   bi = bidx[r];
#pragma unroll
    for (int off = 8; off >= 1; off >>= 1) {
      const float ov = __shfl_xor(bv, off, 32);
      const int   oi = __shfl_xor(bi, off, 32);
      if (ov > bv || (ov == bv && oi < bi)) { bv = ov; bi = oi; }  // lowest idx on tie
    }
    if (mrow == 0) {
      const int row = half * 8 + r;              // 0..15 within wave tile
      lds_idx[w][row] = bi;
      idx_out[n0 + w * 16 + row] = bi;
    }
  }
  __syncthreads();

  // Gather e^T[idx] and write q_ste (== quantized) and quantized; stores are
  // 64B-coalesced per 16-lane half (consecutive hw at fixed feature d).
  const int    myidx = lds_idx[w][mrow];
  const float* erow  = eT + (size_t)myidx * DIM;
  const size_t obase = xbase + (size_t)hw_row;
#pragma unroll 4
  for (int d0 = 0; d0 < DIM; d0 += 2) {
    const int    d = d0 + half;
    const float  v = erow[d];
    const size_t o = obase + (size_t)d * HW;
    qout1[o] = v;
    qout2[o] = v;
  }
}

// ---------------------------------------------------------------------------
extern "C" void kernel_launch(void* const* d_in, const int* in_sizes, int n_in,
                              void* d_out, int out_size, void* d_ws, size_t ws_size,
                              hipStream_t stream) {
  (void)in_sizes; (void)n_in; (void)out_size; (void)ws_size;
  const float* x = (const float*)d_in[0];   // [64,256,32,32] f32
  const float* e = (const float*)d_in[1];   // [256,1024] f32
  float* out = (float*)d_out;

  uint8_t*  ws  = (uint8_t*)d_ws;
  uint32_t* bhi = (uint32_t*)ws;                       // 512 KB B hi fragments
  uint32_t* blo = (uint32_t*)(ws + (512u << 10));      // 512 KB B lo fragments
  float*    eT  = (float*)(ws + (1024u << 10));        // 1 MB  e^T [K][D]
  float*    hn  = (float*)(ws + (2048u << 10));        // 4 KB  0.5*||e_k||^2

  vq_prep_norm_t<<<KEMB / 256, 256, 0, stream>>>(e, eT, hn);
  vq_prep_frags<<<(NKT * NDB * 32) / 256, 256, 0, stream>>>(e, bhi, blo);

  float* q1  = out;                                    // q_ste
  float* q2  = out + (size_t)NROWS * DIM;              // quantized
  int*   idx = (int*)(out + 2 * (size_t)NROWS * DIM);  // indices (int32)
  vq_main<<<NROWS / TILE_M, 256, 0, stream>>>(x, bhi, blo, eT, hn, q1, q2, idx);
}
